// GlobalBlock_12756052869258
// MI455X (gfx1250) — matrix operand order repaired and checked by reading, not compiled, not run on hardware
//
#include <hip/hip_runtime.h>
#include <hip/hip_bf16.h>
#include <math.h>

// ---------------- problem constants (from reference) ----------------
#define DMODEL 1024
#define NHEAD  16
#define HDIM   64
#define SEQ    2048
#define BATCH  2
#define ROWS   (BATCH * SEQ)   // 4096
#define D3     (3 * DMODEL)    // 3072
#define DFF    (4 * DMODEL)    // 4096
#define STRIDE_G 128
#define NGLOB  (SEQ / STRIDE_G) // 16

typedef __bf16 bf16;
typedef __attribute__((ext_vector_type(8)))  __bf16 v8bf;
typedef __attribute__((ext_vector_type(16))) __bf16 v16bf;
typedef __attribute__((ext_vector_type(8)))  float  v8f;

typedef __attribute__((ext_vector_type(4))) unsigned int u32x4;
typedef __attribute__((ext_vector_type(4))) int          i32x4;
typedef __attribute__((ext_vector_type(8))) int          i32x8;

#if defined(__has_builtin)
#if __has_builtin(__builtin_amdgcn_tensor_load_to_lds) && __has_builtin(__builtin_amdgcn_s_wait_tensorcnt)
#define HAVE_TDM 1
#else
#define HAVE_TDM 0
#endif
#else
#define HAVE_TDM 0
#endif

// LDS row stride in elements after TDM padding: 32 bf16 (64B = 16 dwords)
// + 4 dwords pad = 80B = 40 elements.  16B-aligned and bank-conflict-free
// across 16 consecutive rows (20 dwords/row, gcd(20,64) spread).
#define LROW 40

// ---------------- TDM 2-D tile load: global[rows x 32 bf16] -> LDS ----------------
// D# group 0: count=1, lds_addr, 57-bit global addr, type=2 ("image").
// D# group 1: data_size=2B (code 1), pad_enable, pad_interval=16dw (code 3),
//             pad_amount=4dw (code 3), tensor_dim0=stride, tensor_dim1=big,
//             tile_dim0=32, tile_dim1=rows, tensor_dim0_stride=stride.
#if HAVE_TDM
__device__ __forceinline__ void tdm_load_tile(unsigned lds_off, const bf16* gptr,
                                              unsigned rows, unsigned stride_elems) {
  const unsigned long long ga = (unsigned long long)(uintptr_t)gptr;
  u32x4 g0;
  g0.x = 1u;                                             // count=1 (valid user desc)
  g0.y = lds_off;                                        // LDS byte address
  g0.z = (unsigned)ga;                                   // global addr [31:0]
  g0.w = (unsigned)((ga >> 32) & 0x01FFFFFFu) | (2u << 30); // addr[56:32] | type=2

  const unsigned tdim1 = 0x100000u;                      // generous row bound (in-bounds tiles)
  i32x8 g1;
  g1[0] = (int)((1u << 16) | (1u << 20) | (3u << 22) | (3u << 25));
  g1[1] = (int)((stride_elems & 0xFFFFu) << 16);         // tensor_dim0[15:0]
  g1[2] = (int)((stride_elems >> 16) | ((tdim1 & 0xFFFFu) << 16)); // dim0[31:16] | dim1[15:0]
  g1[3] = (int)((tdim1 >> 16) | (32u << 16));            // dim1[31:16] | tile_dim0=32
  g1[4] = (int)(rows & 0xFFFFu);                         // tile_dim1 = rows (tile_dim2=0)
  g1[5] = (int)stride_elems;                             // tensor_dim0_stride[31:0]
  g1[6] = 0;                                             // stride[47:32] | dim1_stride lo
  g1[7] = 0;

  const i32x4 z4 = {0, 0, 0, 0};
#if defined(__clang_major__) && (__clang_major__ >= 23)
  const i32x8 z8 = {0, 0, 0, 0, 0, 0, 0, 0};
  __builtin_amdgcn_tensor_load_to_lds(g0, g1, z4, z4, z8, 0);
#else
  __builtin_amdgcn_tensor_load_to_lds(g0, g1, z4, z4, 0);
#endif
}
#endif

// ---------------- fragment loaders from padded LDS tiles ----------------
// A-matrix 16x32 bf16: lane holds row (lane&15); lane-half hi:
//   elems 0..7 -> K = 8*hi + (0..7) ; elems 8..15 -> K = 16 + 8*hi + (0..7)
__device__ __forceinline__ v16bf lds_frag_a(const bf16* __restrict__ tile,
                                            int row, int lane) {
  const int hi = lane >> 4;
  const bf16* p = tile + (row + (lane & 15)) * LROW + hi * 8;
  union { v16bf f; v8bf h[2]; } u;
  u.h[0] = *(const v8bf*)(p);
  u.h[1] = *(const v8bf*)(p + 16);
  return u.f;
}
// B-matrix 32x16 bf16: lane holds col (lane&15); elems 0..7 -> K = 8*hi + (0..7) ; elems 8..15 -> K = 16 + 8*hi + (0..7)
__device__ __forceinline__ v16bf lds_frag_b(const bf16* __restrict__ tile,
                                            int col, int lane) {
  const int hi = lane >> 4;
  const bf16* p = tile + (col + (lane & 15)) * LROW + hi * 8;
  union { v16bf f; v8bf h[2]; } u;
  u.h[0] = *(const v8bf*)(p);
  u.h[1] = *(const v8bf*)(p + 16);
  return u.f;
}

// ---------------- WMMA GEMM: out[M][N] = A[M][K] @ W[N][K]^T + bias (+epilogue) ----
// Block = 4 waves (128 thr), tile 128M x 64N; per wave 32M x 64N.
// Operands staged to LDS by the Tensor Data Mover, double-buffered.
// EP = 0 : store bf16 ; EP = 1 : exact GELU -> bf16 ; EP = 2 : +res -> f32
template <int EP>
__global__ void __launch_bounds__(128)
gemm_wmma(const bf16* __restrict__ A, const bf16* __restrict__ W,
          const float* __restrict__ bias, const float* __restrict__ res,
          void* __restrict__ outv, int M, int N, int K) {
  __shared__ bf16 sA[2][128 * LROW];
  __shared__ bf16 sB[2][64 * LROW];

  const int lane = threadIdx.x & 31;
  const int wave = threadIdx.x >> 5;
  const int n0   = blockIdx.x * 64;
  const int mblk = blockIdx.y * 128;
  const int mw   = wave * 32;            // wave's row offset inside the block tile

  const v8f vzero = {0.f, 0.f, 0.f, 0.f, 0.f, 0.f, 0.f, 0.f};
  v8f acc[2][4];
#pragma unroll
  for (int i = 0; i < 2; ++i)
#pragma unroll
    for (int j = 0; j < 4; ++j) acc[i][j] = vzero;

  const int nk = K >> 5;                 // K / 32 tiles

#if HAVE_TDM
  if (wave == 0) {
    tdm_load_tile((unsigned)(uintptr_t)&sA[0][0], A + (size_t)mblk * K, 128u, (unsigned)K);
    tdm_load_tile((unsigned)(uintptr_t)&sB[0][0], W + (size_t)n0 * K,   64u,  (unsigned)K);
  }
#endif

  for (int kb = 0; kb < nk; ++kb) {
    const int buf = kb & 1;
#if HAVE_TDM
    if (wave == 0) {
      if (kb + 1 < nk) {   // kick off next tile pair into the other buffer
        const int nb = buf ^ 1;
        tdm_load_tile((unsigned)(uintptr_t)&sA[nb][0],
                      A + (size_t)mblk * K + (kb + 1) * 32, 128u, (unsigned)K);
        tdm_load_tile((unsigned)(uintptr_t)&sB[nb][0],
                      W + (size_t)n0 * K + (kb + 1) * 32, 64u, (unsigned)K);
        __builtin_amdgcn_s_wait_tensorcnt(2);   // current pair done, next pair in flight
      } else {
        __builtin_amdgcn_s_wait_tensorcnt(0);
      }
    }
    __syncthreads();                     // LDS tiles visible to all waves
#else
    // Fallback: cooperative synchronous copy through VGPRs.
    {
      const bf16* ga = A + (size_t)mblk * K + kb * 32;
#pragma unroll
      for (int it = 0; it < 4; ++it) {
        const int idx = it * 128 + threadIdx.x;       // 512 x 16B chunks
        const int r = idx >> 2, c = (idx & 3) * 8;
        *(v8bf*)&sA[buf][r * LROW + c] = *(const v8bf*)(ga + (size_t)r * K + c);
      }
      const bf16* gb = W + (size_t)n0 * K + kb * 32;
#pragma unroll
      for (int it = 0; it < 2; ++it) {
        const int idx = it * 128 + threadIdx.x;       // 256 x 16B chunks
        const int r = idx >> 2, c = (idx & 3) * 8;
        *(v8bf*)&sB[buf][r * LROW + c] = *(const v8bf*)(gb + (size_t)r * K + c);
      }
    }
    __syncthreads();
#endif

    v16bf a[2];
    a[0] = lds_frag_a(&sA[buf][0], mw, lane);
    a[1] = lds_frag_a(&sA[buf][0], mw + 16, lane);
    v16bf b[4];
#pragma unroll
    for (int j = 0; j < 4; ++j) b[j] = lds_frag_b(&sB[buf][0], 16 * j, lane);

#pragma unroll
    for (int i = 0; i < 2; ++i)
#pragma unroll
      for (int j = 0; j < 4; ++j)
        acc[i][j] = __builtin_amdgcn_wmma_f32_16x16x32_bf16(
            false, a[i], false, b[j], (short)0, acc[i][j], false, false);

    __syncthreads();                     // all waves done reading before buffer reuse
  }

  // C/D layout: lane holds col (lane&15); vgpr r holds row r + 8*(lane>>4)
  const int cb = lane & 15;
  const int rb = (lane >> 4) * 8;
#pragma unroll
  for (int i = 0; i < 2; ++i) {
#pragma unroll
    for (int j = 0; j < 4; ++j) {
      const int c = n0 + j * 16 + cb;
      const float bv = bias[c];
#pragma unroll
      for (int r = 0; r < 8; ++r) {
        const int rr = mblk + mw + i * 16 + rb + r;
        const size_t off = (size_t)rr * N + c;
        float v = acc[i][j][r] + bv;
        if (EP == 0) {
          ((bf16*)outv)[off] = (bf16)v;
        } else if (EP == 1) {
          const float t = 0.5f * v * (1.f + erff(v * 0.70710678f));
          ((bf16*)outv)[off] = (bf16)t;
        } else {
          ((float*)outv)[off] = v + res[off];
        }
      }
    }
  }
}

// ---------------- LayerNorm -> bf16 (one block per row) ----------------
__global__ void __launch_bounds__(256)
layernorm_bf16(const float* __restrict__ x, const float* __restrict__ g,
               const float* __restrict__ bta, bf16* __restrict__ out) {
  const int row = blockIdx.x;
  const float* xr = x + (size_t)row * DMODEL;
  float s = 0.f, s2 = 0.f;
  for (int i = threadIdx.x; i < DMODEL; i += 256) {
    const float v = xr[i];
    s += v; s2 += v * v;
  }
#pragma unroll
  for (int o = 16; o >= 1; o >>= 1) {
    s  += __shfl_xor(s,  o, 32);
    s2 += __shfl_xor(s2, o, 32);
  }
  __shared__ float ss[8], ss2[8];
  if ((threadIdx.x & 31) == 0) { ss[threadIdx.x >> 5] = s; ss2[threadIdx.x >> 5] = s2; }
  __syncthreads();
  s = 0.f; s2 = 0.f;
#pragma unroll
  for (int w = 0; w < 8; ++w) { s += ss[w]; s2 += ss2[w]; }
  const float mu   = s * (1.f / DMODEL);
  const float var  = s2 * (1.f / DMODEL) - mu * mu;
  const float rstd = rsqrtf(var + 1e-5f);
  bf16* orow = out + (size_t)row * DMODEL;
  for (int i = threadIdx.x; i < DMODEL; i += 256)
    orow[i] = (bf16)((xr[i] - mu) * rstd * g[i] + bta[i]);
}

// ---------------- stripe-masked attention ----------------
// allowed(i,j) = (i % 128 == 0) || (j % 128 == 0)
//  -> non-global query rows attend exactly the 16 global keys;
//     global query rows attend all 2048 keys.
// One wave per (b,h,t) query row; each lane owns 2 of the 64 head dims.
__global__ void __launch_bounds__(256)
attn_stripe(const bf16* __restrict__ qkv, bf16* __restrict__ o) {
  const int lane = threadIdx.x & 31;
  const int rid  = blockIdx.x * 8 + (threadIdx.x >> 5);
  const int t = rid % SEQ;
  const int h = (rid / SEQ) % NHEAD;
  const int b = rid / (SEQ * NHEAD);

  const int d0 = lane * 2;
  const bf16* qp = qkv + (size_t)(b * SEQ + t) * D3 + h * HDIM;
  const float q0 = (float)qp[d0], q1 = (float)qp[d0 + 1];

  const bool full = (t % STRIDE_G) == 0;
  const int  nk   = full ? SEQ : NGLOB;

  float m = -INFINITY, l = 0.f, o0 = 0.f, o1 = 0.f;
  for (int it = 0; it < nk; ++it) {
    const int j = full ? it : it * STRIDE_G;
    const bf16* kp = qkv + (size_t)(b * SEQ + j) * D3 + DMODEL + h * HDIM;
    float d = q0 * (float)kp[d0] + q1 * (float)kp[d0 + 1];
#pragma unroll
    for (int ofs = 16; ofs >= 1; ofs >>= 1) d += __shfl_xor(d, ofs, 32);
    d *= 0.125f;  // 1/sqrt(64)
    const float mn    = fmaxf(m, d);
    const float alpha = __expf(m - mn);   // first iter: exp(-inf)=0
    const float p     = __expf(d - mn);
    const bf16* vp = qkv + (size_t)(b * SEQ + j) * D3 + 2 * DMODEL + h * HDIM;
    l  = l  * alpha + p;
    o0 = o0 * alpha + p * (float)vp[d0];
    o1 = o1 * alpha + p * (float)vp[d0 + 1];
    m = mn;
  }
  const float inv = 1.f / l;
  bf16* op = o + (size_t)(b * SEQ + t) * DMODEL + h * HDIM;
  op[d0]     = (bf16)(o0 * inv);
  op[d0 + 1] = (bf16)(o1 * inv);
}

// ---------------- f32 -> bf16 convert ----------------
__global__ void __launch_bounds__(256)
cvt_bf16(const float* __restrict__ src, bf16* __restrict__ dst, int n) {
  int i = blockIdx.x * 256 + threadIdx.x;
  const int stride = gridDim.x * 256;
  for (; i < n; i += stride) dst[i] = (bf16)src[i];
}

// ---------------- host launcher ----------------
extern "C" void kernel_launch(void* const* d_in, const int* in_sizes, int n_in,
                              void* d_out, int out_size, void* d_ws, size_t ws_size,
                              hipStream_t stream) {
  (void)in_sizes; (void)n_in; (void)out_size; (void)ws_size;

  const float* x     = (const float*)d_in[0];
  const float* ln1_g = (const float*)d_in[1];
  const float* ln1_b = (const float*)d_in[2];
  const float* w_in  = (const float*)d_in[3];
  const float* b_in  = (const float*)d_in[4];
  const float* w_out = (const float*)d_in[5];
  const float* b_out = (const float*)d_in[6];
  const float* ln2_g = (const float*)d_in[7];
  const float* ln2_b = (const float*)d_in[8];
  const float* w1    = (const float*)d_in[9];
  const float* b1    = (const float*)d_in[10];
  const float* w2    = (const float*)d_in[11];
  const float* b2    = (const float*)d_in[12];
  // d_in[13] = stride (128), baked in.

  char*  ws  = (char*)d_ws;
  size_t off = 0;
  auto carve = [&](size_t bytes) -> void* {
    void* p = ws + off;
    off += (bytes + 255) & ~(size_t)255;
    return p;
  };

  bf16*  w_in_bf  = (bf16*) carve((size_t)D3 * DMODEL * 2);
  bf16*  w_out_bf = (bf16*) carve((size_t)DMODEL * DMODEL * 2);
  bf16*  w1_bf    = (bf16*) carve((size_t)DFF * DMODEL * 2);
  bf16*  w2_bf    = (bf16*) carve((size_t)DMODEL * DFF * 2);
  bf16*  xn_bf    = (bf16*) carve((size_t)ROWS * DMODEL * 2);
  bf16*  qkv_bf   = (bf16*) carve((size_t)ROWS * D3 * 2);
  bf16*  o_bf     = (bf16*) carve((size_t)ROWS * DMODEL * 2);
  float* y_f      = (float*)carve((size_t)ROWS * DMODEL * 4);
  bf16*  yn_bf    = (bf16*) carve((size_t)ROWS * DMODEL * 2);
  bf16*  h1_bf    = (bf16*) carve((size_t)ROWS * DFF * 2);

  // 1) weights -> bf16 (L2-resident thereafter: 24MB << 192MB L2)
  cvt_bf16<<<(D3 * DMODEL) / 256,     256, 0, stream>>>(w_in,  w_in_bf,  D3 * DMODEL);
  cvt_bf16<<<(DMODEL * DMODEL) / 256, 256, 0, stream>>>(w_out, w_out_bf, DMODEL * DMODEL);
  cvt_bf16<<<(DFF * DMODEL) / 256,    256, 0, stream>>>(w1,    w1_bf,    DFF * DMODEL);
  cvt_bf16<<<(DMODEL * DFF) / 256,    256, 0, stream>>>(w2,    w2_bf,    DMODEL * DFF);

  // 2) LN1 -> bf16
  layernorm_bf16<<<ROWS, 256, 0, stream>>>(x, ln1_g, ln1_b, xn_bf);

  // 3) qkv = xn @ w_in^T + b_in   (M=4096, N=3072, K=1024) -> bf16
  gemm_wmma<0><<<dim3(D3 / 64, ROWS / 128), 128, 0, stream>>>(
      xn_bf, w_in_bf, b_in, nullptr, qkv_bf, ROWS, D3, DMODEL);

  // 4) stripe-masked attention -> o (bf16)
  attn_stripe<<<(BATCH * NHEAD * SEQ) / 8, 256, 0, stream>>>(qkv_bf, o_bf);

  // 5) y = x + o @ w_out^T + b_out  (f32)
  gemm_wmma<2><<<dim3(DMODEL / 64, ROWS / 128), 128, 0, stream>>>(
      o_bf, w_out_bf, b_out, x, y_f, ROWS, DMODEL, DMODEL);

  // 6) LN2 -> bf16
  layernorm_bf16<<<ROWS, 256, 0, stream>>>(y_f, ln2_g, ln2_b, yn_bf);

  // 7) h1 = gelu(yn @ w1^T + b1) -> bf16  (M=4096, N=4096, K=1024)
  gemm_wmma<1><<<dim3(DFF / 64, ROWS / 128), 128, 0, stream>>>(
      yn_bf, w1_bf, b1, nullptr, h1_bf, ROWS, DFF, DMODEL);

  // 8) out = x + h1 @ w2^T + b2  (M=4096, N=1024, K=4096) -> f32 d_out
  gemm_wmma<2><<<dim3(DMODEL / 64, ROWS / 128), 128, 0, stream>>>(
      h1_bf, w2_bf, b2, x, (float*)d_out, ROWS, DMODEL, DFF);
}
